// ResidualCosSimVQ_79525614452864
// MI455X (gfx1250) — compile-verified, matrix operand order checked by reading, not run on hardware
//
#include <hip/hip_runtime.h>

// Problem constants (B,N,D,C,Q from reference)
#define BB 8
#define NN 2048
#define DD 512
#define CC 4096
#define QQ 4
#define ROWS (BB * NN)            // 16384
#define SPLITS 4                  // code-dimension splits in argmax kernel

typedef __attribute__((ext_vector_type(16))) __bf16 v16bf;
typedef __attribute__((ext_vector_type(8)))  __bf16 v8bf;
typedef __attribute__((ext_vector_type(8)))  float  v8f;

// ---------------------------------------------------------------- utilities
__device__ __forceinline__ float waveReduceSum(float v) {
#pragma unroll
  for (int m = 16; m >= 1; m >>= 1) v += __shfl_xor(v, m, 32);
  return v;
}

__device__ __forceinline__ v16bf pack2(v8bf lo, v8bf hi) {
  return __builtin_shufflevector(lo, hi, 0, 1, 2, 3, 4, 5, 6, 7,
                                 8, 9, 10, 11, 12, 13, 14, 15);
}

// Fragment load from bf16 row-major memory; p = row base + (lane>>4)*8 offset
// handled by caller via kb. Fragment = K[kb..kb+7] ++ K[kb+16..kb+23].
__device__ __forceinline__ v16bf frag_bf16(const __bf16* p) {
  return pack2(((const v8bf*)p)[0], ((const v8bf*)(p + 16))[0]);
}

// Load a 16-bit A/B fragment (CDNA5 layout) from an f32 row-major matrix.
__device__ __forceinline__ v16bf frag_from_f32(const float* p) {
  float4 a0 = ((const float4*)p)[0];
  float4 a1 = ((const float4*)p)[1];
  float4 b0 = ((const float4*)(p + 16))[0];
  float4 b1 = ((const float4*)(p + 16))[1];
  v16bf f;
  f[0]  = (__bf16)a0.x; f[1]  = (__bf16)a0.y; f[2]  = (__bf16)a0.z; f[3]  = (__bf16)a0.w;
  f[4]  = (__bf16)a1.x; f[5]  = (__bf16)a1.y; f[6]  = (__bf16)a1.z; f[7]  = (__bf16)a1.w;
  f[8]  = (__bf16)b0.x; f[9]  = (__bf16)b0.y; f[10] = (__bf16)b0.z; f[11] = (__bf16)b0.w;
  f[12] = (__bf16)b1.x; f[13] = (__bf16)b1.y; f[14] = (__bf16)b1.z; f[15] = (__bf16)b1.w;
  return f;
}

// ---------------------------------------------------------------- kernel 0
__global__ void k_init(const float* __restrict__ x, float* __restrict__ residual,
                       float* __restrict__ outq, float* __restrict__ loss) {
  size_t i = (size_t)blockIdx.x * blockDim.x + threadIdx.x;
  size_t total = (size_t)ROWS * DD;
  size_t stride = (size_t)gridDim.x * blockDim.x;
  for (size_t t = i; t < total; t += stride) {
    residual[t] = x[t];
    outq[t] = 0.0f;
  }
  if (i < QQ) loss[i] = 0.0f;
}

// ---------------------------------------------------------------- kernel 1
// x_n(bf16) = l2norm(residual) row-wise; one wave per row
__global__ void k_rownorm_bf16(const float* __restrict__ residual,
                               __bf16* __restrict__ xnbf) {
  int row  = (int)((blockIdx.x * blockDim.x + threadIdx.x) >> 5);
  int lane = threadIdx.x & 31;
  if (row >= ROWS) return;
  const float* rp = residual + (size_t)row * DD;
  float4 v[4];
#pragma unroll
  for (int j = 0; j < 4; j++) v[j] = ((const float4*)rp)[lane * 4 + j];
  float ss = 0.f;
#pragma unroll
  for (int j = 0; j < 4; j++)
    ss += v[j].x * v[j].x + v[j].y * v[j].y + v[j].z * v[j].z + v[j].w * v[j].w;
  ss = waveReduceSum(ss);
  float inv = 1.0f / fmaxf(sqrtf(ss), 1e-12f);
  __bf16* op = xnbf + (size_t)row * DD + lane * 16;
#pragma unroll
  for (int j = 0; j < 2; j++) {
    float4 a = v[2 * j], b = v[2 * j + 1];
    v8bf o;
    o[0] = (__bf16)(a.x * inv); o[1] = (__bf16)(a.y * inv);
    o[2] = (__bf16)(a.z * inv); o[3] = (__bf16)(a.w * inv);
    o[4] = (__bf16)(b.x * inv); o[5] = (__bf16)(b.y * inv);
    o[6] = (__bf16)(b.z * inv); o[7] = (__bf16)(b.w * inv);
    ((v8bf*)op)[j] = o;
  }
}

// ---------------------------------------------------------------- kernel 2
// icb = cb @ W^T via WMMA bf16 (f32 inputs converted in-register).
__global__ __launch_bounds__(128) void k_icb_gemm(const float* __restrict__ cb,
                                                  const float* __restrict__ W,
                                                  float* __restrict__ icb) {
  int w = threadIdx.x >> 5, lane = threadIdx.x & 31;
  int rowtile = blockIdx.x >> 3;              // 256 row tiles
  int coltile = (blockIdx.x & 7) * 4 + w;     // 32 col tiles
  int rbase = rowtile * 16, cbase = coltile * 16;
  int lrow = lane & 15, lkoff = (lane >> 4) * 8;
  const float* arow = cb + (size_t)(rbase + lrow) * DD;
  const float* brow = W  + (size_t)(cbase + lrow) * DD;
  v8f acc = {};
#pragma unroll
  for (int kk = 0; kk < 16; kk++) {
    int kb = kk * 32 + lkoff;
    v16bf a = frag_from_f32(arow + kb);
    v16bf b = frag_from_f32(brow + kb);
    acc = __builtin_amdgcn_wmma_f32_16x16x32_bf16(false, a, false, b,
                                                  (short)0, acc, false, false);
  }
  int orow = rbase + (lane < 16 ? 0 : 8);
#pragma unroll
  for (int r = 0; r < 8; r++)
    icb[(size_t)(orow + r) * DD + cbase + lrow] = acc[r];
}

// ---------------------------------------------------------------- kernel 2b
// cb_n (f32 + bf16) = l2norm(icb) row-wise; one wave per row
__global__ void k_cbnorm(const float* __restrict__ icb, float* __restrict__ cbn,
                         __bf16* __restrict__ cbnbf) {
  int row  = (int)((blockIdx.x * blockDim.x + threadIdx.x) >> 5);
  int lane = threadIdx.x & 31;
  if (row >= CC) return;
  const float* rp = icb + (size_t)row * DD;
  float4 v[4];
#pragma unroll
  for (int j = 0; j < 4; j++) v[j] = ((const float4*)rp)[lane * 4 + j];
  float ss = 0.f;
#pragma unroll
  for (int j = 0; j < 4; j++)
    ss += v[j].x * v[j].x + v[j].y * v[j].y + v[j].z * v[j].z + v[j].w * v[j].w;
  ss = waveReduceSum(ss);
  float inv = 1.0f / fmaxf(sqrtf(ss), 1e-12f);
  float* fo = cbn + (size_t)row * DD + lane * 16;
  __bf16* op = cbnbf + (size_t)row * DD + lane * 16;
#pragma unroll
  for (int j = 0; j < 4; j++)
    ((float4*)fo)[j] = make_float4(v[j].x * inv, v[j].y * inv, v[j].z * inv, v[j].w * inv);
#pragma unroll
  for (int j = 0; j < 2; j++) {
    float4 a = v[2 * j], b = v[2 * j + 1];
    v8bf o;
    o[0] = (__bf16)(a.x * inv); o[1] = (__bf16)(a.y * inv);
    o[2] = (__bf16)(a.z * inv); o[3] = (__bf16)(a.w * inv);
    o[4] = (__bf16)(b.x * inv); o[5] = (__bf16)(b.y * inv);
    o[6] = (__bf16)(b.z * inv); o[7] = (__bf16)(b.w * inv);
    ((v8bf*)op)[j] = o;
  }
}

// ---------------------------------------------------------------- kernel 3
// Fused sim = x_n @ cb_n^T + argmax, 2x2 register-blocked (32 rows x 32 codes
// per wave per tile step -> 4 WMMAs per 4 fragment loads = 16 FLOP/B).
// Block = 4 waves * 32 rows = 128 rows staged in 128KB dynamic LDS.
// Code dim split SPLITS ways across blocks; partial argmax written to ws.
__global__ __launch_bounds__(128) void k_argmax(const __bf16* __restrict__ xnbf,
                                                const __bf16* __restrict__ cbnbf,
                                                float* __restrict__ pval,
                                                int* __restrict__ pidx) {
  extern __shared__ __bf16 smem[];       // 128 rows * 512 K * bf16 = 128 KB
  int w = threadIdx.x >> 5, lane = threadIdx.x & 31;
  int rowgroup = blockIdx.x >> 2;        // 128 groups of 128 rows
  int split    = blockIdx.x & (SPLITS - 1);
  int rbase = rowgroup * 128 + w * 32;   // this wave's 32 rows

  // stage wave's 32 rows x 512 K of bf16 x_n into LDS (row-major)
  const v8bf* src = (const v8bf*)(xnbf + (size_t)rbase * DD);
  v8bf* dst = (v8bf*)(smem + w * 32 * DD);
#pragma unroll
  for (int i = 0; i < 64; i++) dst[i * 32 + lane] = src[i * 32 + lane];
  __syncthreads();

  int lrow = lane & 15, lkoff = (lane >> 4) * 8;
  const __bf16* a0base = smem + (w * 32 + lrow) * DD;   // rows rbase+0..15
  const __bf16* a1base = a0base + 16 * DD;              // rows rbase+16..31

  float best0[8], best1[8];
  int   bidx0[8], bidx1[8];
#pragma unroll
  for (int r = 0; r < 8; r++) {
    best0[r] = -3.4e38f; best1[r] = -3.4e38f;
    bidx0[r] = 0;        bidx1[r] = 0;
  }

  int codebase = split * (CC / SPLITS);            // 1024 codes per split
  for (int ct = 0; ct < (CC / SPLITS) / 32; ct++) { // 32 tiles of 32 codes
    int code0 = codebase + ct * 32;
    const __bf16* b0base = cbnbf + (size_t)(code0 + lrow) * DD;  // codes +0..15
    const __bf16* b1base = b0base + 16 * DD;                     // codes +16..31
    if (ct + 1 < 32)
      __builtin_prefetch(b0base + 32 * DD + lkoff, 0, 0);

    v8f acc00 = {}, acc01 = {}, acc10 = {}, acc11 = {};
#pragma unroll 4
    for (int kk = 0; kk < 16; kk++) {
      int kb = kk * 32 + lkoff;
      v16bf a0 = frag_bf16(a0base + kb);
      v16bf a1 = frag_bf16(a1base + kb);
      v16bf b0 = frag_bf16(b0base + kb);
      v16bf b1 = frag_bf16(b1base + kb);
      acc00 = __builtin_amdgcn_wmma_f32_16x16x32_bf16(false, a0, false, b0,
                                                      (short)0, acc00, false, false);
      acc01 = __builtin_amdgcn_wmma_f32_16x16x32_bf16(false, a0, false, b1,
                                                      (short)0, acc01, false, false);
      acc10 = __builtin_amdgcn_wmma_f32_16x16x32_bf16(false, a1, false, b0,
                                                      (short)0, acc10, false, false);
      acc11 = __builtin_amdgcn_wmma_f32_16x16x32_bf16(false, a1, false, b1,
                                                      (short)0, acc11, false, false);
    }
    int c0 = code0 + lrow, c1 = c0 + 16;   // c0 < c1: check c0 first (argmax ties)
#pragma unroll
    for (int r = 0; r < 8; r++) {
      if (acc00[r] > best0[r]) { best0[r] = acc00[r]; bidx0[r] = c0; }
      if (acc01[r] > best0[r]) { best0[r] = acc01[r]; bidx0[r] = c1; }
      if (acc10[r] > best1[r]) { best1[r] = acc10[r]; bidx1[r] = c0; }
      if (acc11[r] > best1[r]) { best1[r] = acc11[r]; bidx1[r] = c1; }
    }
  }

  // butterfly argmax across the 16 lanes sharing each row; write partials
  size_t pbase = (size_t)split * ROWS;
#pragma unroll
  for (int r = 0; r < 8; r++) {
    float v0 = best0[r]; int i0 = bidx0[r];
    float v1 = best1[r]; int i1 = bidx1[r];
#pragma unroll
    for (int m = 1; m < 16; m <<= 1) {
      float ov0 = __shfl_xor(v0, m, 32); int oi0 = __shfl_xor(i0, m, 32);
      float ov1 = __shfl_xor(v1, m, 32); int oi1 = __shfl_xor(i1, m, 32);
      if (ov0 > v0 || (ov0 == v0 && oi0 < i0)) { v0 = ov0; i0 = oi0; }
      if (ov1 > v1 || (ov1 == v1 && oi1 < i1)) { v1 = ov1; i1 = oi1; }
    }
    if (lane == 0) {
      pval[pbase + rbase + r] = v0;       pidx[pbase + rbase + r] = i0;
      pval[pbase + rbase + 16 + r] = v1;  pidx[pbase + rbase + 16 + r] = i1;
    } else if (lane == 16) {
      pval[pbase + rbase + 8 + r] = v0;   pidx[pbase + rbase + 8 + r] = i0;
      pval[pbase + rbase + 24 + r] = v1;  pidx[pbase + rbase + 24 + r] = i1;
    }
  }
}

// ---------------------------------------------------------------- kernel 3b
// Merge per-split argmax partials (splits are ascending code ranges).
__global__ void k_argmax_merge(const float* __restrict__ pval,
                               const int* __restrict__ pidx,
                               int* __restrict__ idxbuf) {
  int row = blockIdx.x * blockDim.x + threadIdx.x;
  if (row >= ROWS) return;
  float bv = -3.4e38f;
  int bi = 0;
#pragma unroll
  for (int s = 0; s < SPLITS; s++) {
    float v = pval[(size_t)s * ROWS + row];
    int ix = pidx[(size_t)s * ROWS + row];
    if (v > bv || (v == bv && ix < bi)) { bv = v; bi = ix; }
  }
  idxbuf[row] = bi;
}

// ---------------------------------------------------------------- kernel 4
// Per-row: gather quant, commit loss, rotation trick, residual/output update.
__global__ void k_update(float* __restrict__ residual, const float* __restrict__ cbn,
                         const int* __restrict__ idxbuf, float* __restrict__ outq,
                         float* __restrict__ outidx, float* __restrict__ loss, int qi) {
  int row  = (int)((blockIdx.x * blockDim.x + threadIdx.x) >> 5);
  int lane = threadIdx.x & 31;
  if (row >= ROWS) return;
  size_t roff = (size_t)row * DD + lane * 16;
  float4 rv[4];
#pragma unroll
  for (int j = 0; j < 4; j++)
    rv[j] = ((const float4*)(residual + (size_t)row * DD))[lane * 4 + j];

  float ss = 0.f;
#pragma unroll
  for (int j = 0; j < 4; j++)
    ss += rv[j].x * rv[j].x + rv[j].y * rv[j].y + rv[j].z * rv[j].z + rv[j].w * rv[j].w;
  ss = waveReduceSum(ss);
  float invr = 1.0f / fmaxf(sqrtf(ss), 1e-12f);

  float xn[16], qv[16];
#pragma unroll
  for (int j = 0; j < 4; j++) {
    xn[4 * j + 0] = rv[j].x * invr; xn[4 * j + 1] = rv[j].y * invr;
    xn[4 * j + 2] = rv[j].z * invr; xn[4 * j + 3] = rv[j].w * invr;
  }

  int code = idxbuf[row];
  const float* qp = cbn + (size_t)code * DD;
#pragma unroll
  for (int j = 0; j < 4; j++) {
    float4 q = ((const float4*)qp)[lane * 4 + j];
    qv[4 * j + 0] = q.x; qv[4 * j + 1] = q.y; qv[4 * j + 2] = q.z; qv[4 * j + 3] = q.w;
  }

  float d2 = 0.f, ns2 = 0.f, nt2 = 0.f;
#pragma unroll
  for (int e = 0; e < 16; e++) {
    float d = xn[e] - qv[e];
    d2 += d * d;
    ns2 += xn[e] * xn[e];
    nt2 += qv[e] * qv[e];
  }
  d2  = waveReduceSum(d2);
  ns2 = waveReduceSum(ns2);
  nt2 = waveReduceSum(nt2);
  if (lane == 0) atomicAdd(loss + qi, d2);

  float ns = sqrtf(ns2), nt = sqrtf(nt2);
  float inv_ns = 1.0f / ns, inv_nt = 1.0f / nt;

  float wv[16];
  float wss = 0.f;
#pragma unroll
  for (int e = 0; e < 16; e++) {
    float t = xn[e] * inv_ns + qv[e] * inv_nt;
    wv[e] = t;
    wss += t * t;
  }
  wss = waveReduceSum(wss);
  float invw = 1.0f / fmaxf(sqrtf(wss), 1e-12f);

  float dew = 0.f;
#pragma unroll
  for (int e = 0; e < 16; e++) { wv[e] *= invw; dew += xn[e] * wv[e]; }
  dew = waveReduceSum(dew);
  float deu = ns2 * inv_ns;
  float scale = nt * inv_ns;

  float* resp = residual + (size_t)row * DD + lane * 16;
  float* outp = outq + roff;
#pragma unroll
  for (int e = 0; e < 16; e++) {
    float r = (xn[e] - 2.0f * dew * wv[e] + 2.0f * deu * (qv[e] * inv_nt)) * scale;
    float base = (e < 4 ? (&rv[0].x)[e] : e < 8 ? (&rv[1].x)[e - 4]
                 : e < 12 ? (&rv[2].x)[e - 8] : (&rv[3].x)[e - 12]);
    resp[e] = base - r;
    outp[e] += r;
  }
  if (lane == 0) outidx[(size_t)row * QQ + qi] = (float)code;
}

// ---------------------------------------------------------------- kernel 5
__global__ void k_loss_finalize(const float* __restrict__ loss,
                                float* __restrict__ outloss) {
  int i = threadIdx.x;
  if (i < QQ) outloss[i] = 1.25f * loss[i] / (float)((size_t)ROWS * DD);
}

// ---------------------------------------------------------------- launch
extern "C" void kernel_launch(void* const* d_in, const int* in_sizes, int n_in,
                              void* d_out, int out_size, void* d_ws, size_t ws_size,
                              hipStream_t stream) {
  const float* x         = (const float*)d_in[0];   // [B,N,D]
  const float* codebooks = (const float*)d_in[1];   // [Q,C,D]
  const float* weights   = (const float*)d_in[2];   // [Q,D,D]

  float* outq    = (float*)d_out;                   // [B*N*D]
  float* outidx  = outq + (size_t)ROWS * DD;        // [B*N*Q] as float
  float* outloss = outidx + (size_t)ROWS * QQ;      // [Q]

  char* ws = (char*)d_ws;
  size_t off = 0;
  float*  residual = (float*)(ws + off);  off += (size_t)ROWS * DD * 4;        // 32 MB
  __bf16* xnbf     = (__bf16*)(ws + off); off += (size_t)ROWS * DD * 2;        // 16 MB
  float*  icb      = (float*)(ws + off);  off += (size_t)CC * DD * 4;          //  8 MB
  float*  cbn      = (float*)(ws + off);  off += (size_t)CC * DD * 4;          //  8 MB
  __bf16* cbnbf    = (__bf16*)(ws + off); off += (size_t)CC * DD * 2;          //  4 MB
  int*    idxbuf   = (int*)(ws + off);    off += (size_t)ROWS * 4;             // 64 KB
  float*  pval     = (float*)(ws + off);  off += (size_t)SPLITS * ROWS * 4;    // 256 KB
  int*    pidx     = (int*)(ws + off);    off += (size_t)SPLITS * ROWS * 4;    // 256 KB
  float*  lossacc  = (float*)(ws + off);  off += 256;

  k_init<<<2048, 256, 0, stream>>>(x, residual, outq, lossacc);

  for (int qi = 0; qi < QQ; qi++) {
    const float* cb = codebooks + (size_t)qi * CC * DD;
    const float* W  = weights   + (size_t)qi * DD * DD;

    // x_n = l2norm(residual) (bf16 copy for WMMA)
    k_rownorm_bf16<<<ROWS / 8, 256, 0, stream>>>(residual, xnbf);
    // icb = cb @ W^T (WMMA bf16)
    k_icb_gemm<<<2048, 128, 0, stream>>>(cb, W, icb);
    // cb_n = l2norm(icb) (f32 + bf16)
    k_cbnorm<<<CC / 8, 256, 0, stream>>>(icb, cbn, cbnbf);
    // fused sim GEMM + argmax, 2x2 blocked, 4-way code split, 128KB LDS
    k_argmax<<<(ROWS / 128) * SPLITS, 128, 128 * 1024, stream>>>(xnbf, cbnbf,
                                                                 pval, pidx);
    k_argmax_merge<<<ROWS / 256, 256, 0, stream>>>(pval, pidx, idxbuf);
    // rotation trick + residual/output/loss update
    k_update<<<ROWS / 8, 256, 0, stream>>>(residual, cbn, idxbuf, outq, outidx,
                                           lossacc, qi);
  }

  k_loss_finalize<<<1, 32, 0, stream>>>(lossacc, outloss);
}
